// CHEAFFusionModule_39694087750164
// MI455X (gfx1250) — matrix operand order repaired and checked by reading, not compiled, not run on hardware
//
#include <hip/hip_runtime.h>
#include <math.h>

// ---------------------------------------------------------------------------
// CDNA5 (gfx1250) implementation of the CHEAF fusion module.
// All 1x1 convs / attention einsums -> LDS-staged WMMA f16 GEMM (f32 accum).
// Block tile 32(M) x 64(N), 4 waves; each wave: 2 WMMAs per K-step sharing
// one B fragment. Branch-free main K loop; guarded zero-padded tail loop.
// ---------------------------------------------------------------------------

typedef __attribute__((ext_vector_type(16))) _Float16 v16h;
typedef __attribute__((ext_vector_type(8)))  _Float16 v8h;
typedef __attribute__((ext_vector_type(8)))  float    v8f;

__device__ __forceinline__ float sigmoidf_(float v) { return 1.f / (1.f + __expf(-v)); }

template<int ACT> __device__ __forceinline__ float apply_act(float v) {
  if (ACT == 1) return v > 0.f ? v : 0.f;       // relu
  if (ACT == 2) return sigmoidf_(v);            // sigmoid
  if (ACT == 3) return v * sigmoidf_(v);        // silu
  return v;
}

__device__ __forceinline__ v16h cat8h(v8h lo, v8h hi) {
  return __builtin_shufflevector(lo, hi, 0, 1, 2, 3, 4, 5, 6, 7,
                                 8, 9, 10, 11, 12, 13, 14, 15);
}

// Y[z][m][n] = act( bn( A[z][m][:] @ X[z][:][n] + bias[m] ) ) + res[z][m][n]
// TRANSB==0: X element (k,n) = X[k*ldX + n]
// TRANSB==1: X element (k,n) = X[n*ldX + k]
// LDS tiles: A 32x32 f16 [m][k], B 32x64 f16 stored transposed [n][k]; both
// with 40-element (80 B) padded rows so fragment reads are 16B-aligned
// ds_load_b128. Fragment layouts per ISA 7.12.2 (16-bit A 16x32 / B 32x16).
template<int TRANSB, int ACT>
__global__ void __launch_bounds__(128) k_wmma_gemm(
    const float* __restrict__ A, long sA, int ldA,
    const float* __restrict__ X, long sX, long ldX,
    const float* __restrict__ bias,
    const float* __restrict__ bnG, const float* __restrict__ bnB,
    const float* __restrict__ res, long sR,
    float* __restrict__ Y, long sY, long ldY,
    int M, long N, int K)
{
  __shared__ _Float16 lA[32 * 40];   // [m][k]
  __shared__ _Float16 lB[64 * 40];   // [n][k] (transposed tile)

  const int tid  = threadIdx.x;
  const int lane = tid & 31;
  const int wave = tid >> 5;
  const long nblk0 = (long)blockIdx.x * 64;
  const int  m0    = blockIdx.y * 32;
  const int  z     = blockIdx.z;

  const float* a_base = A + (long)z * sA;
  const float* x_base = X + (long)z * sX;
  float*       y_base = Y + (long)z * sY;

  // Fragment coordinates:
  //   A: lanes 0-15 row=lane, K {0..7}+{16..23}; lanes 16-31 row=lane-16, K {8..15}+{24..31}
  //   B: lanes 0-15 col=lane, K {0..7}+{16..23}; lanes 16-31 col=lane-16, K {8..15}+{24..31}
  const int rowL   = lane & 15;
  const int klaneA = (lane >> 4) << 3;
  const int colL   = (wave << 4) + (lane & 15);
  const int klaneB = (lane >> 4) << 3;

  v8f acc0 = {}, acc1 = {};

  const bool tileFull = (nblk0 + 64 <= N) && (m0 + 32 <= M);
  const int  kMain    = tileFull ? (K & ~31) : 0;

  int k0 = 0;
  // ======================= branch-free main loop ===========================
  for (; k0 < kMain; k0 += 32) {
    // ---- A tile: 32x32 floats = 256 float4 groups, 2 per thread ----------
#pragma unroll
    for (int it = 0; it < 2; ++it) {
      int idx = tid + it * 128;
      int r   = idx >> 3;            // 0..31
      int kg  = (idx & 7) << 2;      // 0..28
      float4 v = *(const float4*)&a_base[(long)(m0 + r) * ldA + (k0 + kg)];
      _Float16* dst = &lA[r * 40 + kg];
      dst[0] = (_Float16)v.x; dst[1] = (_Float16)v.y;
      dst[2] = (_Float16)v.z; dst[3] = (_Float16)v.w;
    }
    // ---- B tile: 32x64 floats = 512 float4 groups, 4 per thread ----------
    if (TRANSB) {
#pragma unroll
      for (int it = 0; it < 4; ++it) {
        int idx = tid + it * 128;
        int nn  = idx >> 3;          // 0..63
        int kg  = (idx & 7) << 2;    // 0..28
        float4 v = *(const float4*)&x_base[(nblk0 + nn) * ldX + (k0 + kg)];
        _Float16* dst = &lB[nn * 40 + kg];
        dst[0] = (_Float16)v.x; dst[1] = (_Float16)v.y;
        dst[2] = (_Float16)v.z; dst[3] = (_Float16)v.w;
      }
    } else {
#pragma unroll
      for (int it = 0; it < 4; ++it) {
        int idx = tid + it * 128;
        int kk  = idx >> 4;          // 0..31
        int ng  = (idx & 15) << 2;   // 0..60
        float4 v = *(const float4*)&x_base[(long)(k0 + kk) * ldX + (nblk0 + ng)];
        lB[(ng + 0) * 40 + kk] = (_Float16)v.x;
        lB[(ng + 1) * 40 + kk] = (_Float16)v.y;
        lB[(ng + 2) * 40 + kk] = (_Float16)v.z;
        lB[(ng + 3) * 40 + kk] = (_Float16)v.w;
      }
    }
    if (k0 + 64 <= K)  // global_prefetch_b8 for the next K slab
      __builtin_prefetch(TRANSB ? &x_base[nblk0 * ldX + (k0 + 32)]
                                : &x_base[(long)(k0 + 32) * ldX + nblk0], 0, 1);
    __syncthreads();

    v16h af0 = cat8h(*(const v8h*)&lA[rowL * 40 + klaneA],
                     *(const v8h*)&lA[rowL * 40 + klaneA + 16]);
    v16h af1 = cat8h(*(const v8h*)&lA[(16 + rowL) * 40 + klaneA],
                     *(const v8h*)&lA[(16 + rowL) * 40 + klaneA + 16]);
    v16h bf  = cat8h(*(const v8h*)&lB[colL * 40 + klaneB],
                     *(const v8h*)&lB[colL * 40 + klaneB + 16]);
    acc0 = __builtin_amdgcn_wmma_f32_16x16x32_f16(false, af0, false, bf,
                                                  (short)0, acc0, false, false);
    acc1 = __builtin_amdgcn_wmma_f32_16x16x32_f16(false, af1, false, bf,
                                                  (short)0, acc1, false, false);
    __syncthreads();
  }
  // ======================= guarded zero-padded tail ========================
  for (; k0 < K; k0 += 32) {
    for (int idx = tid; idx < 32 * 32; idx += 128) {
      int r = idx >> 5, kk = idx & 31;
      float v = 0.f;
      if (m0 + r < M && k0 + kk < K) v = a_base[(long)(m0 + r) * ldA + (k0 + kk)];
      lA[r * 40 + kk] = (_Float16)v;
    }
    for (int idx = tid; idx < 32 * 64; idx += 128) {
      int kk = idx & 31, nn = idx >> 5;
      float v = 0.f;
      long n = nblk0 + nn;
      if (n < N && k0 + kk < K)
        v = TRANSB ? x_base[n * ldX + (k0 + kk)] : x_base[(long)(k0 + kk) * ldX + n];
      lB[nn * 40 + kk] = (_Float16)v;
    }
    __syncthreads();
    v16h af0 = cat8h(*(const v8h*)&lA[rowL * 40 + klaneA],
                     *(const v8h*)&lA[rowL * 40 + klaneA + 16]);
    v16h af1 = cat8h(*(const v8h*)&lA[(16 + rowL) * 40 + klaneA],
                     *(const v8h*)&lA[(16 + rowL) * 40 + klaneA + 16]);
    v16h bf  = cat8h(*(const v8h*)&lB[colL * 40 + klaneB],
                     *(const v8h*)&lB[colL * 40 + klaneB + 16]);
    acc0 = __builtin_amdgcn_wmma_f32_16x16x32_f16(false, af0, false, bf,
                                                  (short)0, acc0, false, false);
    acc1 = __builtin_amdgcn_wmma_f32_16x16x32_f16(false, af1, false, bf,
                                                  (short)0, acc1, false, false);
    __syncthreads();
  }

  // ---- epilogue: bias -> BN -> act -> residual ------------------------------
  const long colB = nblk0 + colL;
  if (colB >= N) return;
  const int mhi = (lane >> 4) << 3;  // C/D layout: vgpr r -> M=r (+8 hi half-wave)
#pragma unroll
  for (int half = 0; half < 2; ++half) {
    const v8f& acc = half ? acc1 : acc0;
    const int mbase = m0 + half * 16 + mhi;
#pragma unroll
    for (int r = 0; r < 8; ++r) {
      int mm = mbase + r;
      if (mm >= M) continue;
      float v = acc[r];
      if (bias) v += bias[mm];
      if (bnG)  v = v * (bnG[mm] * 0.99999500003749968f) + bnB[mm];  // g/sqrt(1+1e-5)
      v = apply_act<ACT>(v);
      long o = (long)mm * ldY + colB;
      if (res) v += res[(long)z * sR + o];
      y_base[o] = v;
    }
  }
}

// ---- softmax over last dim (one block per row), in place --------------------
__global__ void __launch_bounds__(256) k_softmax_rows(float* __restrict__ x, long N) {
  float* r = x + (long)blockIdx.x * N;
  __shared__ float sd[256];
  int tid = threadIdx.x;
  float m = -3.4e38f;
  for (long i = tid; i < N; i += 256) m = fmaxf(m, r[i]);
  sd[tid] = m; __syncthreads();
  for (int s = 128; s > 0; s >>= 1) { if (tid < s) sd[tid] = fmaxf(sd[tid], sd[tid + s]); __syncthreads(); }
  m = sd[0]; __syncthreads();
  float sum = 0.f;
  for (long i = tid; i < N; i += 256) { float e = __expf(r[i] - m); r[i] = e; sum += e; }
  sd[tid] = sum; __syncthreads();
  for (int s = 128; s > 0; s >>= 1) { if (tid < s) sd[tid] += sd[tid + s]; __syncthreads(); }
  float inv = 1.f / sd[0];
  for (long i = tid; i < N; i += 256) r[i] *= inv;
}

// ---- softmax over channel dim (one thread per pixel), in place --------------
__global__ void k_softmax_chan(float* __restrict__ x, int C, long N, long total) {
  long t = (long)blockIdx.x * blockDim.x + threadIdx.x;
  if (t >= total) return;
  long b = t / N, n = t % N;
  float* base = x + b * (long)C * N + n;
  float m = -3.4e38f;
  for (int c = 0; c < C; ++c) m = fmaxf(m, base[(long)c * N]);
  float s = 0.f;
  for (int c = 0; c < C; ++c) { float e = __expf(base[(long)c * N] - m); base[(long)c * N] = e; s += e; }
  float inv = 1.f / s;
  for (int c = 0; c < C; ++c) base[(long)c * N] *= inv;
}

// ---- depthwise KxK conv, pad=K/2, stride 1 ----------------------------------
template<int ACT>
__global__ void __launch_bounds__(256) k_dwconv(
    const float* __restrict__ x, const float* __restrict__ w, const float* __restrict__ bias,
    float* __restrict__ y, int C, int H, int W, int Ks, long total)
{
  long t = (long)blockIdx.x * blockDim.x + threadIdx.x;
  if (t >= total) return;
  long HWp = (long)H * W;
  long bc  = t / HWp;
  int rem  = (int)(t % HWp);
  int hi = rem / W, wi = rem % W;
  int c = (int)(bc % C);
  const float* xp = x + bc * HWp;
  const float* wp = w + (long)c * Ks * Ks;
  int pad = Ks >> 1;
  float acc = bias[c];
  for (int kh = 0; kh < Ks; ++kh) {
    int hh = hi - pad + kh;
    if ((unsigned)hh >= (unsigned)H) continue;
    for (int kw = 0; kw < Ks; ++kw) {
      int ww = wi - pad + kw;
      if ((unsigned)ww >= (unsigned)W) continue;
      acc += xp[(long)hh * W + ww] * wp[kh * Ks + kw];
    }
  }
  y[t] = apply_act<ACT>(acc);
}

// ---- full 3x3 conv, stride 2, pad 1 -----------------------------------------
__global__ void __launch_bounds__(256) k_conv3s2(
    const float* __restrict__ x, const float* __restrict__ w, const float* __restrict__ bias,
    float* __restrict__ y, int Ci, int Co, int Hi, int Wi, long total)
{
  long t = (long)blockIdx.x * blockDim.x + threadIdx.x;
  if (t >= total) return;
  int Ho = Hi >> 1, Wo = Wi >> 1;
  long HWo = (long)Ho * Wo;
  long bo = t / HWo;
  int rem = (int)(t % HWo);
  int ho = rem / Wo, wo = rem % Wo;
  int o = (int)(bo % Co);
  int b = (int)(bo / Co);
  float acc = bias[o];
  for (int ci = 0; ci < Ci; ++ci) {
    const float* xp = x + ((long)b * Ci + ci) * Hi * Wi;
    const float* wp = w + ((long)o * Ci + ci) * 9;
    for (int kh = 0; kh < 3; ++kh) {
      int hh = 2 * ho - 1 + kh;
      if ((unsigned)hh >= (unsigned)Hi) continue;
      for (int kw = 0; kw < 3; ++kw) {
        int ww = 2 * wo - 1 + kw;
        if ((unsigned)ww >= (unsigned)Wi) continue;
        acc += xp[(long)hh * Wi + ww] * wp[kh * 3 + kw];
      }
    }
  }
  y[t] = acc;
}

// ---- dst = other + bilinear_x2(src)  (half-pixel centers) -------------------
__global__ void __launch_bounds__(256) k_up2_add(
    const float* __restrict__ src, const float* __restrict__ other, float* __restrict__ dst,
    int Hs, int Wsz, long total)
{
  long t = (long)blockIdx.x * blockDim.x + threadIdx.x;
  if (t >= total) return;
  int Hd = Hs * 2, Wd = Wsz * 2;
  long HWd = (long)Hd * Wd;
  long bc = t / HWd;
  int rem = (int)(t % HWd);
  int hd = rem / Wd, wd = rem % Wd;
  const float* sp = src + bc * (long)Hs * Wsz;
  float sy = 0.5f * hd - 0.25f, sx = 0.5f * wd - 0.25f;
  int y0 = (int)floorf(sy), x0 = (int)floorf(sx);
  float fy = sy - (float)y0, fx = sx - (float)x0;
  int y0c = min(max(y0, 0), Hs - 1),  y1c = min(max(y0 + 1, 0), Hs - 1);
  int x0c = min(max(x0, 0), Wsz - 1), x1c = min(max(x0 + 1, 0), Wsz - 1);
  float v00 = sp[(long)y0c * Wsz + x0c], v01 = sp[(long)y0c * Wsz + x1c];
  float v10 = sp[(long)y1c * Wsz + x0c], v11 = sp[(long)y1c * Wsz + x1c];
  float v = (1.f - fy) * ((1.f - fx) * v00 + fx * v01) + fy * ((1.f - fx) * v10 + fx * v11);
  dst[t] = other[t] + v;
}

// ---- global average pool over HW (one block per (b,c)) ----------------------
__global__ void __launch_bounds__(256) k_avgpool(const float* __restrict__ x,
                                                 float* __restrict__ out, long N) {
  const float* r = x + (long)blockIdx.x * N;
  __shared__ float sd[256];
  float s = 0.f;
  for (long i = threadIdx.x; i < N; i += 256) s += r[i];
  sd[threadIdx.x] = s; __syncthreads();
  for (int st = 128; st > 0; st >>= 1) { if (threadIdx.x < st) sd[threadIdx.x] += sd[threadIdx.x + st]; __syncthreads(); }
  if (threadIdx.x == 0) out[blockIdx.x] = sd[0] / (float)N;
}

// ---- SE FCs: cw = sigmoid(Wc2 @ relu(Wc1 @ avg + bc1) + bc2), per batch -----
__global__ void __launch_bounds__(128) k_se_fc(
    const float* __restrict__ avg, const float* __restrict__ Wc1, const float* __restrict__ bc1,
    const float* __restrict__ Wc2, const float* __restrict__ bc2,
    float* __restrict__ cw, int C, int Cr)
{
  __shared__ float a[256];
  __shared__ float tt[16];
  int b = blockIdx.x;
  for (int c = threadIdx.x; c < C; c += blockDim.x) a[c] = avg[(long)b * C + c];
  __syncthreads();
  for (int r = threadIdx.x; r < Cr; r += blockDim.x) {
    float s = bc1[r];
    for (int c = 0; c < C; ++c) s += Wc1[(long)r * C + c] * a[c];
    tt[r] = s > 0.f ? s : 0.f;
  }
  __syncthreads();
  for (int o = threadIdx.x; o < C; o += blockDim.x) {
    float s = bc2[o];
    for (int r = 0; r < Cr; ++r) s += Wc2[(long)o * Cr + r] * tt[r];
    cw[(long)b * C + o] = sigmoidf_(s);
  }
}

// ---- sw[b,n] = sigmoid( Ws2 . s[b,:,n] + bs2 ) ------------------------------
__global__ void __launch_bounds__(256) k_sw_sig(
    const float* __restrict__ s, const float* __restrict__ Ws2, const float* __restrict__ bs2,
    float* __restrict__ sw, int Cr, long N, long total)
{
  long t = (long)blockIdx.x * blockDim.x + threadIdx.x;
  if (t >= total) return;
  long b = t / N, n = t % N;
  const float* sp = s + b * (long)Cr * N + n;
  float acc = bs2[0];
  for (int r = 0; r < Cr; ++r) acc += Ws2[r] * sp[(long)r * N];
  sw[t] = sigmoidf_(acc);
}

// ---- elementwise helpers ----------------------------------------------------
__global__ void __launch_bounds__(256) k_mul_chan(const float* __restrict__ x,
    const float* __restrict__ cw, float* __restrict__ y, long HW, long total) {
  long t = (long)blockIdx.x * blockDim.x + threadIdx.x;
  if (t >= total) return;
  long bc = t / HW;
  y[t] = x[t] * cw[bc];                       // cw laid out [B*C]
}
__global__ void __launch_bounds__(256) k_mul_spatial(const float* __restrict__ x,
    const float* __restrict__ sw, float* __restrict__ y, int C, long HW, long total) {
  long t = (long)blockIdx.x * blockDim.x + threadIdx.x;
  if (t >= total) return;
  long hw = t % HW;
  long b  = t / ((long)C * HW);
  y[t] = x[t] * sw[b * HW + hw];
}
__global__ void __launch_bounds__(256) k_add(const float* __restrict__ a,
    const float* __restrict__ b, float* __restrict__ y, long total) {
  long t = (long)blockIdx.x * blockDim.x + threadIdx.x;
  if (t >= total) return;
  y[t] = a[t] + b[t];
}
// joint[0:B]=x ; joint[B:2B][c][:]=proto[b][c]
__global__ void __launch_bounds__(256) k_make_joint(const float* __restrict__ x,
    const float* __restrict__ proto, float* __restrict__ joint, long HW,
    long S, long total) {
  long t = (long)blockIdx.x * blockDim.x + threadIdx.x;
  if (t >= total) return;
  if (t < S) { joint[t] = x[t]; return; }
  long u = t - S;
  long bc = u / HW;
  joint[t] = proto[bc];                       // proto laid out [B*C]
}

// ---------------------------------------------------------------------------
// Host-side orchestration
// ---------------------------------------------------------------------------
static inline unsigned nblk(long n) { return (unsigned)((n + 255) / 256); }

static void launch_gemm(hipStream_t stream, int transB, int act,
                        const float* A, long sA, int ldA,
                        const float* X, long sX, long ldX,
                        const float* bias, const float* bnG, const float* bnB,
                        const float* res, long sR,
                        float* Y, long sY, long ldY,
                        int M, long N, int K, int batches)
{
  dim3 block(128);
  dim3 grid((unsigned)((N + 63) / 64), (unsigned)((M + 31) >> 5), (unsigned)batches);
#define GEMM_CASE(TB, AC) \
  k_wmma_gemm<TB, AC><<<grid, block, 0, stream>>>(A, sA, ldA, X, sX, ldX, bias, bnG, bnB, \
                                                  res, sR, Y, sY, ldY, M, N, K)
  if (transB == 0) { if (act == 0) GEMM_CASE(0, 0); else if (act == 1) GEMM_CASE(0, 1); else GEMM_CASE(0, 3); }
  else             { if (act == 0) GEMM_CASE(1, 0); else if (act == 1) GEMM_CASE(1, 1); else GEMM_CASE(1, 3); }
#undef GEMM_CASE
}

struct GateP { const float *Wc1, *bc1, *Wc2, *bc2, *Ws1, *bs1, *g, *be, *Ws2, *bs2; };

// ChannelSpatialGate: out = (x*cw) * sw
static void run_gate(hipStream_t stream, const float* x, const GateP& g, int C, int Hd,
                     float* xc, float* sbuf, float* swb, float* avgb, float* cwb, float* out)
{
  const int Bb = 8;
  long N = (long)Hd * Hd;
  int Cr = C / 16;
  long Sf = (long)Bb * C * N;
  k_avgpool<<<Bb * C, 256, 0, stream>>>(x, avgb, N);
  k_se_fc<<<Bb, 128, 0, stream>>>(avgb, g.Wc1, g.bc1, g.Wc2, g.bc2, cwb, C, Cr);
  k_mul_chan<<<nblk(Sf), 256, 0, stream>>>(x, cwb, xc, N, Sf);
  // s = relu(bn(Ws1 @ xc + bs1))  [B, Cr, N]
  launch_gemm(stream, 0, 1, g.Ws1, 0, C, xc, (long)C * N, N, g.bs1, g.g, g.be,
              nullptr, 0, sbuf, (long)Cr * N, N, Cr, N, C, Bb);
  k_sw_sig<<<nblk((long)Bb * N), 256, 0, stream>>>(sbuf, g.Ws2, g.bs2, swb, Cr, N, (long)Bb * N);
  k_mul_spatial<<<nblk(Sf), 256, 0, stream>>>(xc, swb, out, C, N, Sf);
}

extern "C" void kernel_launch(void* const* d_in, const int* in_sizes, int n_in,
                              void* d_out, int out_size, void* d_ws, size_t ws_size,
                              hipStream_t stream)
{
  (void)in_sizes; (void)n_in; (void)out_size; (void)ws_size;
  const int CHs[4] = {32, 64, 128, 256};
  const int HWs[4] = {160, 80, 40, 20};
  const int Bb = 8;

  // ---- unpack inputs (setup_inputs insertion order, params pytree in key order)
  int ti = 0;
  auto IN = [&]() -> const float* { return (const float*)d_in[ti++]; };
  const float *f[4], *proto[4];
  for (int i = 0; i < 4; ++i) { f[i] = IN(); proto[i] = IN(); }
  struct ElaP { const float *Wk, *bk, *Wq, *bq, *Wv, *bv, *Wr, *br; } E[4];
  for (int i = 0; i < 4; ++i) {
    E[i].Wk = IN(); E[i].bk = IN(); E[i].Wq = IN(); E[i].bq = IN();
    E[i].Wv = IN(); E[i].bv = IN(); E[i].Wr = IN(); E[i].br = IN();
  }
  struct SlcP { const float *Ws, *bs, *Wl, *bl; } SL[4];
  for (int i = 0; i < 4; ++i) { SL[i].Ws = IN(); SL[i].bs = IN(); SL[i].Wl = IN(); SL[i].bl = IN(); }
  GateP G[4];
  for (int i = 0; i < 4; ++i) {
    G[i].Wc1 = IN(); G[i].bc1 = IN(); G[i].Wc2 = IN(); G[i].bc2 = IN();
    G[i].Ws1 = IN(); G[i].bs1 = IN(); G[i].g = IN(); G[i].be = IN();
    G[i].Ws2 = IN(); G[i].bs2 = IN();
  }
  struct WB { const float *W, *b; } TD[3], BU[3], OP[4];
  for (int i = 0; i < 3; ++i) { TD[i].W = IN(); TD[i].b = IN(); }
  for (int i = 0; i < 3; ++i) { BU[i].W = IN(); BU[i].b = IN(); }
  struct FuseP { const float *Wdw, *bdw, *Wpw, *bpw, *g, *be; } FU[4];
  for (int i = 0; i < 4; ++i) {
    FU[i].Wdw = IN(); FU[i].bdw = IN(); FU[i].Wpw = IN(); FU[i].bpw = IN();
    FU[i].g = IN(); FU[i].be = IN();
  }
  for (int i = 0; i < 4; ++i) { OP[i].W = IN(); OP[i].b = IN(); }

  // ---- workspace (bump allocator)
  size_t off = 0;
  auto alloc = [&](long nflt) -> float* {
    float* p = (float*)((char*)d_ws + off);
    off += (((size_t)nflt * 4) + 255) & ~(size_t)255;
    return p;
  };
  long HW2[4], S[4];
  for (int i = 0; i < 4; ++i) { HW2[i] = (long)HWs[i] * HWs[i]; S[i] = (long)Bb * CHs[i] * HW2[i]; }

  float* fused[4]; for (int i = 0; i < 4; ++i) fused[i] = alloc(S[i]);
  float* tdb[4];   for (int i = 0; i < 4; ++i) tdb[i]   = alloc(S[i]);
  float* bub[4];   for (int i = 0; i < 4; ++i) bub[i]   = alloc(S[i]);
  float* avgb = alloc(Bb * 256);
  float* cwb  = alloc(Bb * 256);
  long S2max = 2 * S[0];
  float* scrJ = alloc(S2max);              // joint [2B,C,N]
  float* scrK = alloc(S2max);              // k / gate xc scratch
  float* scrQ = alloc(S2max);              // q
  float* scrV = alloc(S2max);              // v / slc output
  float* scrA = alloc(S2max);              // attention output
  float* scrT = alloc(S[0]);               // misc
  float* scrU = alloc(S[0]);               // misc
  float* ctxb = alloc((long)16 * 256 * 256);
  float* sgate = alloc(S[0] / 16 + 256);   // [B, Cr, N]
  float* swb   = alloc((long)Bb * HW2[0]); // [B, N]

  // ===================== Stage 1: ELA + short-long conv ======================
  for (int i = 0; i < 4; ++i) {
    int C = CHs[i]; long N = HW2[i]; long Sl = S[i]; int B2 = 2 * Bb;
    long cs = (long)C * N;
    k_make_joint<<<nblk(2 * Sl), 256, 0, stream>>>(f[i], proto[i], scrJ, N, Sl, 2 * Sl);
    // projections over the joint batch
    launch_gemm(stream, 0, 0, E[i].Wk, 0, C, scrJ, cs, N, E[i].bk, nullptr, nullptr,
                nullptr, 0, scrK, cs, N, C, N, C, B2);
    launch_gemm(stream, 0, 0, E[i].Wq, 0, C, scrJ, cs, N, E[i].bq, nullptr, nullptr,
                nullptr, 0, scrQ, cs, N, C, N, C, B2);
    launch_gemm(stream, 0, 0, E[i].Wv, 0, C, scrJ, cs, N, E[i].bv, nullptr, nullptr,
                nullptr, 0, scrV, cs, N, C, N, C, B2);
    k_softmax_rows<<<B2 * C, 256, 0, stream>>>(scrK, N);                       // over n
    k_softmax_chan<<<nblk((long)B2 * N), 256, 0, stream>>>(scrQ, C, N, (long)B2 * N); // over c
    // ctx^T[vc][kc] = sum_n v[vc][n] * k[kc][n]   (trans-B GEMM, K-dim = N)
    launch_gemm(stream, 1, 0, scrV, cs, (int)N, scrK, cs, N, nullptr, nullptr, nullptr,
                nullptr, 0, ctxb, (long)C * C, C, C, (long)C, (int)N, B2);
    // att = ctx^T @ q
    launch_gemm(stream, 0, 0, ctxb, (long)C * C, C, scrQ, cs, N, nullptr, nullptr, nullptr,
                nullptr, 0, scrA, cs, N, C, N, C, B2);
    // ela_out = Wr @ att + br + joint   (first B batches only)
    launch_gemm(stream, 0, 0, E[i].Wr, 0, C, scrA, cs, N, E[i].br, nullptr, nullptr,
                scrJ, cs, scrT, cs, N, C, N, C, Bb);
    // short-long conv branch: silu(dw3(x)) then dw7
    k_dwconv<3><<<nblk(Sl), 256, 0, stream>>>(f[i], SL[i].Ws, SL[i].bs, scrU, C, HWs[i], HWs[i], 3, Sl);
    k_dwconv<0><<<nblk(Sl), 256, 0, stream>>>(scrU, SL[i].Wl, SL[i].bl, scrV, C, HWs[i], HWs[i], 7, Sl);
    k_add<<<nblk(Sl), 256, 0, stream>>>(scrT, scrV, fused[i], Sl);
  }

  // ===================== Stage 2: pyramid ====================================
  float* tdp[4]; tdp[3] = fused[3];
  for (int i = 0; i < 3; ++i) tdp[i] = tdb[i];
  // top-down
  for (int i = 2; i >= 0; --i) {
    int C = CHs[i], Cn = CHs[i + 1];
    long Nn = HW2[i + 1];
    // 1x1 lateral on the coarser level, then gated add with 2x bilinear upsample
    launch_gemm(stream, 0, 0, TD[i].W, 0, Cn, tdp[i + 1], (long)Cn * Nn, Nn, TD[i].b,
                nullptr, nullptr, nullptr, 0, scrT, (long)C * Nn, Nn, C, Nn, Cn, Bb);
    run_gate(stream, fused[i], G[i], C, HWs[i], scrK, sgate, swb, avgb, cwb, scrU);
    k_up2_add<<<nblk(S[i]), 256, 0, stream>>>(scrT, scrU, tdb[i], HWs[i + 1], HWs[i + 1], S[i]);
  }
  // bottom-up
  float* bup[4]; bup[0] = tdp[0];
  for (int i = 1; i < 4; ++i) {
    k_conv3s2<<<nblk(S[i]), 256, 0, stream>>>(bup[i - 1], BU[i - 1].W, BU[i - 1].b, scrT,
                                              CHs[i - 1], CHs[i], HWs[i - 1], HWs[i - 1], S[i]);
    run_gate(stream, tdp[i], G[i], CHs[i], HWs[i], scrK, sgate, swb, avgb, cwb, scrU);
    k_add<<<nblk(S[i]), 256, 0, stream>>>(scrT, scrU, bub[i], S[i]);
    bup[i] = bub[i];
  }

  // ===================== Stage 3: fuse + output projection ===================
  float* outp = (float*)d_out;
  long ooff = 0;
  for (int i = 0; i < 4; ++i) {
    int C = CHs[i]; long N = HW2[i];
    k_dwconv<0><<<nblk(S[i]), 256, 0, stream>>>(bup[i], FU[i].Wdw, FU[i].bdw, scrT,
                                                C, HWs[i], HWs[i], 3, S[i]);
    // refined = silu(bn(Wpw @ y + bpw)) + fused[i]
    launch_gemm(stream, 0, 3, FU[i].Wpw, 0, C, scrT, (long)C * N, N, FU[i].bpw,
                FU[i].g, FU[i].be, fused[i], (long)C * N, scrU, (long)C * N, N, C, N, C, Bb);
    // out = Wout @ refined + bout
    launch_gemm(stream, 0, 0, OP[i].W, 0, C, scrU, (long)C * N, N, OP[i].b,
                nullptr, nullptr, nullptr, 0, outp + ooff, (long)C * N, N, C, N, C, Bb);
    ooff += S[i];
  }
}
